// Reflect_59605556134004
// MI455X (gfx1250) — matrix operand order repaired and checked by reading, not compiled, not run on hardware
//
#include <hip/hip_runtime.h>

// ---------------------------------------------------------------------------
// Single-head attention, B=4, S=2048, D=1024, gfx1250 (MI455X).
//   out = softmax(Q K^T / sqrt(D)) V,  Q=xWq, K=xWk, V=xWv
// ~120 GFLOP vs ~80 MB unique data -> compute-bound -> bf16 WMMA everywhere.
// Mask is a no-op: reference adds NEG uniformly along the key axis for masked
// query rows, and softmax is invariant to per-row constant shifts.
//
// Data movement plan:
//   k0: x fp32 -> bf16 once (removes per-tile converts from GEMM inner loops)
//   k1: Q = xb@Wq, K = xb@Wk (row-major bf16), V^T (= [b][d][s] bf16) so that
//       the P@V B-fragments are contiguous 32B loads.  W chunk staged in LDS
//       transposed so B-fragments are two ds_load_b128 (no scalar u16 loads).
//   k2: per 16-query-row block: QK^T (WMMA) -> fp32 logits in LDS (128 KB),
//       row softmax -> unnormalized bf16 P in LDS, O = P@V^T (WMMA), row
//       1/sum folded into the writeback.  224 KB LDS < 320 KB/WGP.
// ---------------------------------------------------------------------------

typedef __attribute__((ext_vector_type(16))) __bf16 v16bf;
typedef __attribute__((ext_vector_type(8)))  __bf16 v8bf;
typedef __attribute__((ext_vector_type(8)))  float  v8f;
typedef __attribute__((ext_vector_type(4)))  float  v4f;

constexpr int Bn = 4;
constexpr int Sn = 2048;
constexpr int Dn = 1024;
constexpr float SCALE = 0.03125f;  // 1/sqrt(1024)

__device__ __forceinline__ v8f zero_v8f() {
  v8f z;
#pragma unroll
  for (int i = 0; i < 8; ++i) z[i] = 0.0f;
  return z;
}

__device__ __forceinline__ v8f wmma_bf16(v16bf a, v16bf b, v8f c) {
  return __builtin_amdgcn_wmma_f32_16x16x32_bf16(false, a, false, b,
                                                 (short)0, c, false, false);
}

// A fragment (16x32 bf16) from a row-major bf16 tile at (row0,k0); ld = stride.
// Lane L: row = L%16; K = kb..kb+7 and kb+16..kb+23, kb = 8*(L>=16).
__device__ __forceinline__ v16bf frag_a_bf16(const __bf16* tile, int ld, int lane) {
  const int r = lane & 15, kb = (lane >> 4) << 3;
  const __bf16* p = tile + r * ld + kb;
  v8bf lo = *(const v8bf*)(p);
  v8bf hi = *(const v8bf*)(p + 16);
  v16bf f;
#pragma unroll
  for (int i = 0; i < 8; ++i) { f[i] = lo[i]; f[i + 8] = hi[i]; }
  return f;
}

// B fragment (32x16) where the k-axis is contiguous in memory:
// element (k,n) = tile[n*ld + k]  (i.e. B = M^T for row-major M).
__device__ __forceinline__ v16bf frag_b_kcontig(const __bf16* tile, int ld, int lane) {
  const int n = lane & 15, koff = (lane >> 4) << 3;
  const __bf16* p = tile + n * ld + koff;
  v8bf lo = *(const v8bf*)(p);
  v8bf hi = *(const v8bf*)(p + 16);
  v16bf f;
#pragma unroll
  for (int i = 0; i < 8; ++i) { f[i] = lo[i]; f[i + 8] = hi[i]; }
  return f;
}

// ---------------------------------------------------------------------------
// Kernel 0: convert x (fp32) -> xb (bf16), 8 elements/thread, vectorized.
// ---------------------------------------------------------------------------
__global__ void cvt_x_kernel(const float* __restrict__ x, __bf16* __restrict__ xb) {
  const size_t i = ((size_t)blockIdx.x * 256 + threadIdx.x) * 8;
  v4f a0 = *(const v4f*)(x + i);
  v4f a1 = *(const v4f*)(x + i + 4);
  v8bf o;
#pragma unroll
  for (int j = 0; j < 4; ++j) { o[j] = (__bf16)a0[j]; o[4 + j] = (__bf16)a1[j]; }
  *(v8bf*)(xb + i) = o;
}

// ---------------------------------------------------------------------------
// Kernel 1: projections.  C[8192,1024] = xb[8192,1024] @ W[1024,1024].
// blockIdx.z: 0 -> Q (row-major), 1 -> K (row-major), 2 -> V^T ([b][d][s]).
// 256 threads = 8 waves, each wave owns 4 M-tiles of one 16-col N-tile.
// W chunk staged in LDS *transposed* ([n][k]) so each lane's B fragment is
// two contiguous ds_load_b128.
// ---------------------------------------------------------------------------
__global__ void qkv_gemm_kernel(const __bf16* __restrict__ xb,
                                const float* __restrict__ Wq,
                                const float* __restrict__ Wk,
                                const float* __restrict__ Wv,
                                __bf16* __restrict__ Qo,
                                __bf16* __restrict__ Ko,
                                __bf16* __restrict__ Vt) {
  const float* W = (blockIdx.z == 0) ? Wq : (blockIdx.z == 1) ? Wk : Wv;

  __shared__ __bf16 sBt[16 * 32];  // W chunk transposed: [n][k]

  const int lane = threadIdx.x & 31;
  const int wave = threadIdx.x >> 5;
  const int col0 = blockIdx.y * 16;
  const int rowbase = blockIdx.x * 512 + wave * 64;

  v8f acc[4];
#pragma unroll
  for (int t = 0; t < 4; ++t) acc[t] = zero_v8f();

  for (int k0 = 0; k0 < Dn; k0 += 32) {
    // stage 32x16 fp32 W chunk into LDS as bf16, transposed to [n][k]
    for (int e = threadIdx.x; e < 512; e += 256) {
      const int nn = e & 15, kk = e >> 4;          // global reads coalesced in n
      sBt[nn * 32 + kk] = (__bf16)W[(size_t)(k0 + kk) * Dn + col0 + nn];
    }
    __syncthreads();

    v16bf bfrag = frag_b_kcontig(sBt, 32, lane);
#pragma unroll
    for (int t = 0; t < 4; ++t) {
      v16bf afrag = frag_a_bf16(xb + (size_t)(rowbase + t * 16) * Dn + k0, Dn, lane);
      acc[t] = wmma_bf16(afrag, bfrag, acc[t]);
    }
    __syncthreads();
  }

  const int n = lane & 15, rb = (lane >> 4) * 8;
  if (blockIdx.z < 2) {
    __bf16* O = (blockIdx.z == 0) ? Qo : Ko;
#pragma unroll
    for (int t = 0; t < 4; ++t)
#pragma unroll
      for (int j = 0; j < 8; ++j)
        O[(size_t)(rowbase + t * 16 + rb + j) * Dn + col0 + n] = (__bf16)acc[t][j];
  } else {
    // V^T: Vt[b][d][s]; m = b*Sn + s
#pragma unroll
    for (int t = 0; t < 4; ++t)
#pragma unroll
      for (int j = 0; j < 8; ++j) {
        const int m = rowbase + t * 16 + rb + j;
        const int bb = m >> 11, s = m & (Sn - 1);
        Vt[((size_t)bb * Dn + col0 + n) * Sn + s] = (__bf16)acc[t][j];
      }
  }
}

// ---------------------------------------------------------------------------
// Kernel 2: fused attention per 16-query-row block (256 threads, 8 waves).
// ---------------------------------------------------------------------------
__global__ void attn_kernel(const __bf16* __restrict__ Qg,
                            const __bf16* __restrict__ Kg,
                            const __bf16* __restrict__ Vt,
                            float* __restrict__ out) {
  extern __shared__ char smem[];
  float*  sLogit = (float*)smem;                                   // 16 x Sn f32
  __bf16* sP     = (__bf16*)(smem + (size_t)16 * Sn * 4);          // 16 x Sn bf16
  __bf16* sQ     = (__bf16*)(smem + (size_t)16 * Sn * 4 + (size_t)16 * Sn * 2);
  __shared__ float sRed[256];
  __shared__ float sRowMax[16];
  __shared__ float sRowInv[16];

  const int lane = threadIdx.x & 31;
  const int wave = threadIdx.x >> 5;
  const int b    = blockIdx.y;
  const int row0 = blockIdx.x * 16;

  const __bf16* Qb  = Qg + (size_t)b * Sn * Dn;
  const __bf16* Kb  = Kg + (size_t)b * Sn * Dn;
  const __bf16* Vtb = Vt + (size_t)b * Dn * Sn;

  // stage the 16 x Dn Q block into LDS
  for (int e = threadIdx.x * 8; e < 16 * Dn; e += 256 * 8)
    *(v8bf*)(sQ + e) = *(const v8bf*)(Qb + (size_t)row0 * Dn + e);
  __syncthreads();

  // ---- Phase 1: logits = Q K^T (each wave: 16 col tiles, 2 groups of 8) ----
  for (int g = 0; g < 2; ++g) {
    v8f acc[8];
#pragma unroll
    for (int t = 0; t < 8; ++t) acc[t] = zero_v8f();

    for (int k0 = 0; k0 < Dn; k0 += 32) {
      v16bf a = frag_a_bf16(sQ + k0, Dn, lane);
      if (k0 + 32 < Dn) {   // prefetch next K chunk (global_prefetch_b8)
        const int c0p = (wave * 16 + g * 8) * 16;
        __builtin_prefetch(Kb + (size_t)(c0p + (lane & 15)) * Dn + k0 + 32, 0, 1);
      }
#pragma unroll
      for (int t = 0; t < 8; ++t) {
        const int c0 = (wave * 16 + g * 8 + t) * 16;
        v16bf bb = frag_b_kcontig(Kb + (size_t)c0 * Dn + k0, Dn, lane);
        acc[t] = wmma_bf16(a, bb, acc[t]);
      }
    }
    const int n = lane & 15, rb = (lane >> 4) * 8;
#pragma unroll
    for (int t = 0; t < 8; ++t) {
      const int c0 = (wave * 16 + g * 8 + t) * 16;
#pragma unroll
      for (int j = 0; j < 8; ++j)
        sLogit[(rb + j) * Sn + c0 + n] = acc[t][j];
    }
  }
  __syncthreads();

  // ---- Phase 2: row softmax (scale folded; unnormalized exp -> bf16 P) -----
  {
    const int row  = threadIdx.x >> 4;   // 16 rows x 16 segments
    const int seg  = threadIdx.x & 15;
    const int cbeg = seg * (Sn / 16);

    float m = -3.0e38f;
    for (int c = cbeg; c < cbeg + Sn / 16; ++c)
      m = fmaxf(m, sLogit[row * Sn + c] * SCALE);
    sRed[threadIdx.x] = m;
    __syncthreads();
    if (seg == 0) {
      float mm = sRed[row * 16];
      for (int i = 1; i < 16; ++i) mm = fmaxf(mm, sRed[row * 16 + i]);
      sRowMax[row] = mm;
    }
    __syncthreads();

    const float rm = sRowMax[row];
    float ssum = 0.0f;
    for (int c = cbeg; c < cbeg + Sn / 16; ++c) {
      const float e = __expf(sLogit[row * Sn + c] * SCALE - rm);
      sP[row * Sn + c] = (__bf16)e;
      ssum += e;
    }
    sRed[threadIdx.x] = ssum;
    __syncthreads();
    if (seg == 0) {
      float tt = 0.0f;
      for (int i = 0; i < 16; ++i) tt += sRed[row * 16 + i];
      sRowInv[row] = 1.0f / tt;
    }
    __syncthreads();
  }

  // ---- Phase 3: O = P @ V via V^T (contiguous B fragments) -----------------
  {
    v8f acc[8];
#pragma unroll
    for (int t = 0; t < 8; ++t) acc[t] = zero_v8f();

    for (int z0 = 0; z0 < Sn; z0 += 32) {
      v16bf a = frag_a_bf16(sP + z0, Sn, lane);
#pragma unroll
      for (int t = 0; t < 8; ++t) {
        const int c0 = (wave * 8 + t) * 16;   // output column tile (d)
        v16bf bb = frag_b_kcontig(Vtb + (size_t)c0 * Sn + z0, Sn, lane);
        acc[t] = wmma_bf16(a, bb, acc[t]);
      }
    }
    const int n = lane & 15, rb = (lane >> 4) * 8;
#pragma unroll
    for (int t = 0; t < 8; ++t) {
      const int c0 = (wave * 8 + t) * 16;
#pragma unroll
      for (int j = 0; j < 8; ++j) {
        const int row = rb + j;
        out[((size_t)b * Sn + row0 + row) * Dn + c0 + n] = acc[t][j] * sRowInv[row];
      }
    }
  }
}

// ---------------------------------------------------------------------------
extern "C" void kernel_launch(void* const* d_in, const int* in_sizes, int n_in,
                              void* d_out, int out_size, void* d_ws, size_t ws_size,
                              hipStream_t stream) {
  const float* x  = (const float*)d_in[0];
  const float* Wq = (const float*)d_in[1];
  const float* Wk = (const float*)d_in[2];
  const float* Wv = (const float*)d_in[3];
  // d_in[4] (mask) unused: adds a per-row constant -> softmax-invariant.
  (void)in_sizes; (void)n_in; (void)out_size; (void)ws_size;

  const size_t NE = (size_t)Bn * Sn * Dn;
  __bf16* Qw = (__bf16*)d_ws;      // [B,S,D]
  __bf16* Kw = Qw + NE;            // [B,S,D]
  __bf16* Vt = Kw + NE;            // [B,D,S]  (transposed V)
  __bf16* xb = Vt + NE;            // [B,S,D]  (x in bf16)

  // Kernel 0: one-time fp32 -> bf16 conversion of x.
  cvt_x_kernel<<<dim3((unsigned)(NE / (256 * 8))), 256, 0, stream>>>(x, xb);

  // Kernel 1: Q/K/V projections (z selects weight + output layout).
  dim3 g1((Bn * Sn) / 512, Dn / 16, 3);
  qkv_gemm_kernel<<<g1, 256, 0, stream>>>(xb, Wq, Wk, Wv, Qw, Kw, Vt);

  // Kernel 2: fused attention, one workgroup per 16 query rows per batch.
  const size_t smem = (size_t)16 * Sn * 4 + (size_t)16 * Sn * 2 + (size_t)16 * Dn * 2;
  (void)hipFuncSetAttribute((const void*)attn_kernel,
                            hipFuncAttributeMaxDynamicSharedMemorySize, (int)smem);
  dim3 g2(Sn / 16, Bn);
  attn_kernel<<<g2, 256, smem, stream>>>(Qw, Kw, Vt, (float*)d_out);
}
